// NerModel_68401649156201
// MI455X (gfx1250) — compile-verified, matrix operand order checked by reading, not run on hardware
//
#include <hip/hip_runtime.h>
#include <hip/hip_bf16.h>

typedef __attribute__((ext_vector_type(16))) _Float16 v16h;
typedef __attribute__((ext_vector_type(8)))  float    v8f;

#define D2   1536
#define TCLS 50
#define NPAD 64
#define NROWS 16384      // B*S = 32*512
#define KCHUNKS 48       // 1536 / 32
#define ROWS_PER_BLOCK 64
#define BLOCK_THREADS  128   // 4 waves * 16 rows = 64 rows/block -> 256 blocks

// workspace layout (bytes)
#define HIST_OFF   0        // 64 ints
#define EXPV_OFF   256      // 64 floats
#define ZV_OFF     512      // 64 floats
#define ACC_OFF    768      // 1 float
#define WPACK_OFF  1024     // 48*4*32*16 halfs = 196608 B

__global__ __launch_bounds__(256) void k0_zero_ws(char* ws) {
    int t = threadIdx.x;
    if (t < 64) ((int*)(ws + HIST_OFF))[t] = 0;
    if (t == 64) ((float*)(ws + ACC_OFF))[0] = 0.0f;
}

__global__ __launch_bounds__(256) void k1_hist(const int* __restrict__ tags, char* ws) {
    int i = blockIdx.x * blockDim.x + threadIdx.x;
    if (i < NROWS) {
        int g = tags[i];
        atomicAdd((int*)(ws + HIST_OFF) + g, 1);
    }
}

__global__ __launch_bounds__(64) void k2_tagconst(const float* __restrict__ score, char* ws) {
    int t = threadIdx.x;
    if (t < TCLS) {
        int s = ((const int*)(ws + HIST_OFF))[t];
        float expo = 1.0f - (float)s / 512.0f;
        float val  = powf(score[t], expo);
        float ev   = expf(val);
        ((float*)(ws + EXPV_OFF))[t] = ev;
        ((float*)(ws + ZV_OFF))[t]   = ev + 49.0f * expf(1.0f);
    }
}

// Pack W [1536 x 50] row-major fp32 into f16 B-fragments.
// B fragment (32x16, f16) per wave-lane layout (mirror of ISA A layout):
//   lane l (N = ntile*16 + (l&15)), base = (l<16)?0:8
//   halfs e=0..7  -> K = kc*32 + base + e
//   halfs e=8..15 -> K = kc*32 + 16 + base + (e-8)
__global__ __launch_bounds__(256) void k3_packW(const float* __restrict__ W, char* ws) {
    int idx = blockIdx.x * blockDim.x + threadIdx.x;         // 98304 total
    if (idx >= KCHUNKS * 4 * 32 * 16) return;
    int e    = idx & 15;
    int lane = (idx >> 4) & 31;
    int nt   = (idx >> 9) & 3;
    int kc   = idx >> 11;
    int base = (lane < 16) ? 0 : 8;
    int K = kc * 32 + ((e < 8) ? (base + e) : (16 + base + (e - 8)));
    int N = nt * 16 + (lane & 15);
    float v = (N < TCLS) ? W[(size_t)K * TCLS + N] : 0.0f;
    ((_Float16*)(ws + WPACK_OFF))[idx] = (_Float16)v;
}

__global__ __launch_bounds__(BLOCK_THREADS) void k4_main(const float* __restrict__ x,
                                                         const int*   __restrict__ tags,
                                                         const float* __restrict__ b,
                                                         char* ws) {
    __shared__ float lds_logits[ROWS_PER_BLOCK * NPAD];      // 16 KB

    const _Float16* wpack = (const _Float16*)(ws + WPACK_OFF);
    const float*    expv  = (const float*)(ws + EXPV_OFF);
    const float*    Zv    = (const float*)(ws + ZV_OFF);

    int tid  = threadIdx.x;
    int wave = tid >> 5;          // 0..3
    int lane = tid & 31;
    int l15  = lane & 15;
    int base = (lane < 16) ? 0 : 8;

    int rbase = blockIdx.x * ROWS_PER_BLOCK + wave * 16;     // wave's 16 rows
    int arow  = rbase + l15;                                 // this lane's A row

    v8f acc0 = {}, acc1 = {}, acc2 = {}, acc3 = {};

    const float* xr = x + (size_t)arow * D2;

    for (int kc = 0; kc < KCHUNKS; ++kc) {
        int k0 = kc * 32 + base;
        float4 f0 = *(const float4*)(xr + k0);
        float4 f1 = *(const float4*)(xr + k0 + 4);
        float4 f2 = *(const float4*)(xr + k0 + 16);
        float4 f3 = *(const float4*)(xr + k0 + 20);

        v16h A;
        A[0]=(_Float16)f0.x;  A[1]=(_Float16)f0.y;  A[2]=(_Float16)f0.z;  A[3]=(_Float16)f0.w;
        A[4]=(_Float16)f1.x;  A[5]=(_Float16)f1.y;  A[6]=(_Float16)f1.z;  A[7]=(_Float16)f1.w;
        A[8]=(_Float16)f2.x;  A[9]=(_Float16)f2.y;  A[10]=(_Float16)f2.z; A[11]=(_Float16)f2.w;
        A[12]=(_Float16)f3.x; A[13]=(_Float16)f3.y; A[14]=(_Float16)f3.z; A[15]=(_Float16)f3.w;

        const v16h* wp = (const v16h*)(wpack + ((size_t)(kc * 4) * 32 + lane) * 16);
        v16h B0 = wp[0 * 32];     // stride between ntiles is 32 lanes * 16 halfs
        v16h B1 = wp[1 * 32];
        v16h B2 = wp[2 * 32];
        v16h B3 = wp[3 * 32];

        acc0 = __builtin_amdgcn_wmma_f32_16x16x32_f16(false, A, false, B0, (short)0, acc0, false, false);
        acc1 = __builtin_amdgcn_wmma_f32_16x16x32_f16(false, A, false, B1, (short)0, acc1, false, false);
        acc2 = __builtin_amdgcn_wmma_f32_16x16x32_f16(false, A, false, B2, (short)0, acc2, false, false);
        acc3 = __builtin_amdgcn_wmma_f32_16x16x32_f16(false, A, false, B3, (short)0, acc3, false, false);
    }

    // Scatter C tiles to LDS per ISA layout: VGPR r -> M = r (lanes 0-15) / r+8 (lanes 16-31)
    int rl_base = wave * 16 + ((lane < 16) ? 0 : 8);
    #pragma unroll
    for (int r = 0; r < 8; ++r) {
        int rl = rl_base + r;
        lds_logits[rl * NPAD + 0  + l15] = acc0[r];
        lds_logits[rl * NPAD + 16 + l15] = acc1[r];
        lds_logits[rl * NPAD + 32 + l15] = acc2[r];
        lds_logits[rl * NPAD + 48 + l15] = acc3[r];
    }
    __syncthreads();

    // One thread per row: softmax stats + losses (closed-form soft-target CE)
    if (tid < ROWS_PER_BLOCK) {
        int grow = blockIdx.x * ROWS_PER_BLOCK + tid;
        const float* lrow = &lds_logits[tid * NPAD];

        float m = -3.4e38f, sum_l = 0.0f;
        #pragma unroll 10
        for (int t = 0; t < TCLS; ++t) {
            float l = lrow[t] + b[t];
            m = fmaxf(m, l);
            sum_l += l;
        }
        float sumexp = 0.0f, pe = 0.0f;
        #pragma unroll 10
        for (int t = 0; t < TCLS; ++t) {
            float d = (lrow[t] + b[t]) - m;
            float e = expf(d);
            sumexp += e;
            pe     += e * d;
        }
        float lse    = logf(sumexp);
        float S_logp = sum_l - (float)TCLS * (m + lse);      // sum_t logp_t
        float loss2  = lse - pe / sumexp;                    // -sum p*logp

        int   g      = tags[grow];
        float logp_g = (lrow[g] + b[g]) - m - lse;
        const float E1 = 2.718281828459045f;
        float loss1  = -(expv[g] * logp_g + E1 * (S_logp - logp_g)) / Zv[g];

        float c = 0.8f * loss1 + 0.2f * loss2;
        atomicAdd((float*)(ws + ACC_OFF), c);
    }
}

__global__ void k5_final(char* ws, float* out) {
    out[0] = ((float*)(ws + ACC_OFF))[0] * (1.0f / (float)NROWS);
}

extern "C" void kernel_launch(void* const* d_in, const int* in_sizes, int n_in,
                              void* d_out, int out_size, void* d_ws, size_t ws_size,
                              hipStream_t stream) {
    const float* x     = (const float*)d_in[0];   // [32,512,1536] f32
    const int*   tags  = (const int*)d_in[1];     // [32,512] int32
    const float* score = (const float*)d_in[2];   // [50] f32
    const float* W     = (const float*)d_in[3];   // [1536,50] f32
    const float* b     = (const float*)d_in[4];   // [50] f32
    float* out = (float*)d_out;
    char*  ws  = (char*)d_ws;

    k0_zero_ws<<<1, 256, 0, stream>>>(ws);
    k1_hist<<<NROWS / 256, 256, 0, stream>>>(tags, ws);
    k2_tagconst<<<1, 64, 0, stream>>>(score, ws);
    k3_packW<<<(KCHUNKS * 4 * 32 * 16 + 255) / 256, 256, 0, stream>>>(W, ws);
    k4_main<<<NROWS / ROWS_PER_BLOCK, BLOCK_THREADS, 0, stream>>>(x, tags, b, ws);
    k5_final<<<1, 1, 0, stream>>>(ws, out);
}
